// ClassQueryHead_45887430590915
// MI455X (gfx1250) — matrix operand order repaired and checked by reading, not compiled, not run on hardware
//
#include <hip/hip_runtime.h>

// ---------------------------------------------------------------- types
typedef __bf16 bf16;
typedef bf16  bf16x8  __attribute__((ext_vector_type(8)));
typedef bf16  bf16x16 __attribute__((ext_vector_type(16)));
typedef float f32x8   __attribute__((ext_vector_type(8)));

union BV16 { bf16x16 v; bf16x8 h[2]; };

// ---------------------------------------------------------------- consts
constexpr int H    = 1024;
constexpr int NH   = 16;
constexpr int DH   = 64;
constexpr int CQ   = 64;     // class queries per graph
constexpr int BG   = 64;     // graphs
constexpr int NNODE= 32000;
constexpr int FF   = 4096;
constexpr int RQ   = BG * CQ;        // 4096 query rows
constexpr int NPAD = 34048;          // 32000 + 64*32 (per-graph 32-aligned bases)
constexpr float EPSLN = 1e-5f;
constexpr float SCALE = 0.125f;      // 1/sqrt(64)

#define DEVI __device__ __forceinline__

DEVI bf16 f2bf(float f) {
    union { float f; unsigned u; } a{f};
    unsigned r = a.u + 0x7fffu + ((a.u >> 16) & 1u);   // round-nearest-even
    unsigned short hsh = (unsigned short)(r >> 16);
    bf16 b; __builtin_memcpy(&b, &hsh, 2); return b;
}

DEVI f32x8 wmma_bf(bf16x16 a, bf16x16 b, f32x8 c) {
    // D = A(16x32 bf16) * B(32x16 bf16) + C(16x16 f32)
    return __builtin_amdgcn_wmma_f32_16x16x32_bf16(false, a, false, b, (short)0, c, false, false);
}

// ------------------- CDNA5 async global->LDS copy (guarded; falls back to ld+ds_store)
#if defined(__AMDGCN__) && __has_builtin(__builtin_amdgcn_global_load_async_to_lds_b128)
#define ASYNC_LDS 1
typedef int v4i __attribute__((vector_size(16)));
#endif

DEVI void cp16(const bf16* g, bf16* l) {
#if defined(ASYNC_LDS)
    // param0: AS1 (global) v4i*   param1: AS3 (LDS) v4i*   then imm offset, imm cpol.
    // Integer round-trips avoid addrspacecast restrictions; flat-LDS low 32 bits are the
    // LDS offset (aperture layout), flat-global bits are the global address.
    __builtin_amdgcn_global_load_async_to_lds_b128(
        (__attribute__((address_space(1))) v4i*)(unsigned long long)g,
        (__attribute__((address_space(3))) v4i*)(unsigned)(unsigned long long)l,
        0, 0);
#else
    *(bf16x8*)l = *(const bf16x8*)g;
#endif
}

DEVI void async_wait0() {
#if defined(ASYNC_LDS)
#if __has_builtin(__builtin_amdgcn_s_wait_asynccnt)
    __builtin_amdgcn_s_wait_asynccnt(0);
#else
    asm volatile("s_wait_asynccnt 0" ::: "memory");
#endif
#endif
}

// ---------------------------------------------------------------- setup kernels
__global__ void k_offsets(const int* __restrict__ batch, int n,
                          int* __restrict__ offs, int* __restrict__ cnt,
                          int* __restrict__ aoff, int* __restrict__ sbase,
                          int* __restrict__ scnt) {
    int t = threadIdx.x;
    if (t <= BG) {               // lower_bound(batch, t)
        int lo = 0, hi = n;
        while (lo < hi) { int mid = (lo + hi) >> 1; if (batch[mid] < t) lo = mid + 1; else hi = mid; }
        offs[t] = lo;
    }
    __syncthreads();
    if (t == 0) {
        aoff[0] = 0;
        for (int b = 0; b < BG; ++b) {
            int c = offs[b + 1] - offs[b];
            cnt[b] = c;
            aoff[b + 1] = aoff[b] + ((c + 31) & ~31);   // 32-aligned compacted base
        }
    }
    if (t < BG) { sbase[t] = t * CQ; scnt[t] = CQ; }
}

__global__ void k_rmap(const int* __restrict__ batch, int n,
                       const int* __restrict__ offs, const int* __restrict__ aoff,
                       int* __restrict__ rmap) {
    int i = blockIdx.x * blockDim.x + threadIdx.x;
    if (i < n) { int b = batch[i]; rmap[i] = aoff[b] + (i - offs[b]); }
}

__global__ void k_cvt(const float* __restrict__ in, bf16* __restrict__ out, long n) {
    long i = (long)blockIdx.x * blockDim.x + threadIdx.x;
    if (i < n) out[i] = f2bf(in[i]);
}

// Wt[n*K + k] = W[k*Nc + n]  (transpose weights for WMMA B operand)
__global__ void k_cvtT(const float* __restrict__ W, bf16* __restrict__ Wt, int K, int Nc) {
    long i = (long)blockIdx.x * blockDim.x + threadIdx.x;
    if (i < (long)K * Nc) {
        int nn = (int)(i / K);
        int k  = (int)(i % K);
        Wt[i] = f2bf(W[(long)k * Nc + nn]);
    }
}

__global__ void k_bcast(const float* __restrict__ cq, float* __restrict__ x) {
    long i = (long)blockIdx.x * blockDim.x + threadIdx.x;
    if (i < (long)RQ * H) {
        int hcol = (int)(i % H);
        int c    = (int)((i / H) % CQ);
        x[i] = cq[c * H + hcol];
    }
}

// ---------------------------------------------------------------- GEMM (bf16 WMMA, f32 acc)
// Block = 128 threads (4 waves) computing a 128x64 tile of C = act(A @ Wt^T + bias) (+Res).
// Each wave owns 32 rows (two 16-row WMMA tiles). The shared 64x32 B tile is staged into
// LDS (double-buffered) with GLOBAL_LOAD_ASYNC_TO_LDS_B128 when available.
// OM: 0 = f32 row-major, 1 = bf16 row-major, 2 = bf16 transposed (ld = ldt)
template <int OM, bool RELU>
__global__ __launch_bounds__(128) void k_gemm(
        const bf16* __restrict__ A, const bf16* __restrict__ Wt,
        const float* __restrict__ bias, const float* __restrict__ Res,
        void* __restrict__ Cout, const int* __restrict__ rmap,
        int M, int Nn, int K, int ldt) {
    constexpr int BSTR = 40;                          // padded LDS stride per column (elems)
    __shared__ __align__(16) bf16 Bs[2 * 64 * BSTR];  // 10 KB

    const int tid  = threadIdx.x;
    const int lane = tid & 31;
    const int wv   = tid >> 5;
    const int ln15 = lane & 15, hihalf = lane >> 4;
    const int n0 = blockIdx.x * 64;
    const int m0 = blockIdx.y * 128 + wv * 32;

    // cooperative stage of one 64col x 32k B tile (4 KB) into LDS buffer `buf`
    auto stage = [&](int buf, int k0) {
#pragma unroll
        for (int i = 0; i < 2; ++i) {
            int c   = i * 128 + tid;                  // 256 chunks of 16 B
            int col = c >> 2;
            int q   = c & 3;
            const bf16* g = Wt + (long)(n0 + col) * K + k0 + q * 8;
            bf16*       l = Bs + buf * (64 * BSTR) + col * BSTR + q * 8;
            cp16(g, l);
        }
    };

    const bf16* arow0 = A + (long)(m0 + ln15) * K;
    const bf16* arow1 = A + (long)(m0 + 16 + ln15) * K;

    f32x8 acc[2][4] = {};
    stage(0, 0);
    for (int k0 = 0; k0 < K; k0 += 32) {
        const int cur = (k0 >> 5) & 1;
        async_wait0();
        __syncthreads();                              // B tile visible to all waves
        if (k0 + 32 < K) stage(cur ^ 1, k0 + 32);     // fill other buffer in flight

        BV16 a0, a1;
        a0.h[0] = *(const bf16x8*)(arow0 + k0 + hihalf * 8);
        a0.h[1] = *(const bf16x8*)(arow0 + k0 + 16 + hihalf * 8);
        a1.h[0] = *(const bf16x8*)(arow1 + k0 + hihalf * 8);
        a1.h[1] = *(const bf16x8*)(arow1 + k0 + 16 + hihalf * 8);
        __builtin_prefetch((const void*)(arow0 + k0 + 256), 0, 1);
#pragma unroll
        for (int t = 0; t < 4; ++t) {
            const bf16* bp = Bs + cur * (64 * BSTR) + (t * 16 + ln15) * BSTR + hihalf * 8;
            BV16 bm;
            bm.h[0] = *(const bf16x8*)(bp);
            bm.h[1] = *(const bf16x8*)(bp + 16);
            acc[0][t] = wmma_bf(a0.v, bm.v, acc[0][t]);
            acc[1][t] = wmma_bf(a1.v, bm.v, acc[1][t]);
        }
    }

#pragma unroll
    for (int mt = 0; mt < 2; ++mt) {
#pragma unroll
        for (int r = 0; r < 8; ++r) {
            int m  = m0 + mt * 16 + r + 8 * hihalf;
            int om = rmap ? rmap[m] : m;
#pragma unroll
            for (int t = 0; t < 4; ++t) {
                int n = n0 + t * 16 + ln15;
                float v = acc[mt][t][r];
                if (bias) v += bias[n];
                if (Res)  v += Res[(long)m * Nn + n];
                if (RELU) v = fmaxf(v, 0.f);
                if constexpr (OM == 0)      ((float*)Cout)[(long)om * Nn + n] = v;
                else if constexpr (OM == 1) ((bf16*)Cout)[(long)om * Nn + n]  = f2bf(v);
                else                        ((bf16*)Cout)[(long)n * ldt + om] = f2bf(v);
            }
        }
    }
}

// ---------------------------------------------------------------- attention (flash-style, WMMA)
// One wave per (graph b, head, query-half mh): 32 queries x all keys of graph b.
// Q:[RQ,H] bf16, Kc:[*,H] bf16 (rows at compacted base), Vt:[H,ldv] bf16 (transposed).
__global__ __launch_bounds__(32) void k_attn(
        const bf16* __restrict__ Q, const bf16* __restrict__ Kc, const bf16* __restrict__ Vt,
        const int* __restrict__ kbase, const int* __restrict__ kcnt,
        float* __restrict__ O, int ldv) {
    __shared__ __align__(32) bf16 pls[32 * 32];   // P tile, row-major [32 q][32 keys]

    const int lane = threadIdx.x;
    const int ln15 = lane & 15, hihalf = lane >> 4;
    const int mh = blockIdx.x, head = blockIdx.y, b = blockIdx.z;
    const int qrow0 = b * CQ + mh * 32;
    const int hoff  = head * DH;
    const int base  = kbase[b];
    const int cnt   = kcnt[b];
    const int ntiles = (cnt + 31) >> 5;

    // Q in A-operand layout: 2 row-tiles x 2 K-steps (dh=64)
    bf16x16 qa[2][2];
#pragma unroll
    for (int mt = 0; mt < 2; ++mt) {
        const bf16* qp = Q + (long)(qrow0 + mt * 16 + ln15) * H + hoff;
#pragma unroll
        for (int ks = 0; ks < 2; ++ks) {
            BV16 t;
            t.h[0] = *(const bf16x8*)(qp + ks * 32 + hihalf * 16);
            t.h[1] = *(const bf16x8*)(qp + ks * 32 + hihalf * 16 + 8);
            qa[mt][ks] = t.v;
        }
    }

    float mx[2][8], sm[2][8];
#pragma unroll
    for (int mt = 0; mt < 2; ++mt)
#pragma unroll
        for (int r = 0; r < 8; ++r) { mx[mt][r] = -3.0e38f; sm[mt][r] = 0.f; }

    int key0 = 0;
    auto computeS = [&](int mt, int nt) -> f32x8 {  // 16q x 16k score tile (scaled+masked)
        f32x8 s = {};
        const bf16* kp = Kc + (long)(base + key0 + nt * 16 + ln15) * H + hoff + hihalf * 16;
#pragma unroll
        for (int ks = 0; ks < 2; ++ks)
            s = wmma_bf(qa[mt][ks], *(const bf16x16*)(kp + ks * 32), s);
        bool oob = (key0 + nt * 16 + ln15) >= cnt;
#pragma unroll
        for (int r = 0; r < 8; ++r) s[r] = oob ? -1e30f : s[r] * SCALE;
        return s;
    };

    // pass 1: online row max + sum-exp
    for (int kt = 0; kt < ntiles; ++kt) {
        key0 = kt * 32;
#pragma unroll
        for (int mt = 0; mt < 2; ++mt) {
            f32x8 s0 = computeS(mt, 0);
            f32x8 s1 = computeS(mt, 1);
#pragma unroll
            for (int r = 0; r < 8; ++r) {
                float v = fmaxf(s0[r], s1[r]);
                for (int off = 8; off > 0; off >>= 1) v = fmaxf(v, __shfl_xor(v, off, 32));
                float mo = mx[mt][r], mn = fmaxf(mo, v);
                float e = __expf(s0[r] - mn) + __expf(s1[r] - mn);
                for (int off = 8; off > 0; off >>= 1) e += __shfl_xor(e, off, 32);
                sm[mt][r] = sm[mt][r] * __expf(mo - mn) + e;
                mx[mt][r] = mn;
            }
        }
    }

    // pass 2: P = exp(s-m)/l  -> LDS relayout -> O += P @ V
    f32x8 oacc[2][4] = {};
    for (int kt = 0; kt < ntiles; ++kt) {
        key0 = kt * 32;
        __syncthreads();
#pragma unroll
        for (int mt = 0; mt < 2; ++mt) {
            f32x8 s0 = computeS(mt, 0);
            f32x8 s1 = computeS(mt, 1);
#pragma unroll
            for (int r = 0; r < 8; ++r) {
                float rs = 1.f / (sm[mt][r] + 1e-30f);
                int row = mt * 16 + r + 8 * hihalf;
                pls[row * 32 + ln15]      = f2bf(__expf(s0[r] - mx[mt][r]) * rs);
                pls[row * 32 + 16 + ln15] = f2bf(__expf(s1[r] - mx[mt][r]) * rs);
            }
        }
        __syncthreads();
        bf16x16 pa[2];
#pragma unroll
        for (int mt = 0; mt < 2; ++mt) {
            const bf16* pp = pls + (mt * 16 + ln15) * 32;
            BV16 t;
            t.h[0] = *(const bf16x8*)(pp + hihalf * 16);
            t.h[1] = *(const bf16x8*)(pp + hihalf * 16 + 8);
            pa[mt] = t.v;
        }
#pragma unroll
        for (int dt = 0; dt < 4; ++dt) {
            const bf16* vp = Vt + (long)(hoff + dt * 16 + ln15) * ldv + base + key0 + hihalf * 16;
            bf16x16 bv = *(const bf16x16*)vp;
#pragma unroll
            for (int mt = 0; mt < 2; ++mt) oacc[mt][dt] = wmma_bf(pa[mt], bv, oacc[mt][dt]);
        }
    }

#pragma unroll
    for (int mt = 0; mt < 2; ++mt)
#pragma unroll
        for (int dt = 0; dt < 4; ++dt)
#pragma unroll
            for (int r = 0; r < 8; ++r)
                O[(long)(qrow0 + mt * 16 + r + 8 * hihalf) * H + hoff + dt * 16 + ln15] = oacc[mt][dt][r];
}

// ---------------------------------------------------------------- layernorm (f32 in -> bf16 out)
__global__ void k_ln(const float* __restrict__ X, const float* __restrict__ g,
                     const float* __restrict__ bb, bf16* __restrict__ Y) {
    int row = blockIdx.x, tid = threadIdx.x;
    const float* xr = X + (long)row * H;
    float s = 0.f, s2 = 0.f;
    for (int c = tid; c < H; c += 256) { float v = xr[c]; s += v; s2 += v * v; }
    __shared__ float sh[256], sh2[256];
    sh[tid] = s; sh2[tid] = s2; __syncthreads();
    for (int o = 128; o > 0; o >>= 1) {
        if (tid < o) { sh[tid] += sh[tid + o]; sh2[tid] += sh2[tid + o]; }
        __syncthreads();
    }
    float mean = sh[0] * (1.f / H);
    float var  = sh2[0] * (1.f / H) - mean * mean;
    float inv  = rsqrtf(var + EPSLN);
    for (int c = tid; c < H; c += 256)
        Y[(long)row * H + c] = f2bf((xr[c] - mean) * inv * g[c] + bb[c]);
}

// final: out[row] = LN(x_row) . w + b
__global__ void k_out(const float* __restrict__ X, const float* __restrict__ g,
                      const float* __restrict__ bb, const float* __restrict__ w,
                      const float* __restrict__ ob, float* __restrict__ out) {
    int row = blockIdx.x, tid = threadIdx.x;
    const float* xr = X + (long)row * H;
    float s = 0.f, s2 = 0.f;
    for (int c = tid; c < H; c += 256) { float v = xr[c]; s += v; s2 += v * v; }
    __shared__ float sh[256], sh2[256];
    sh[tid] = s; sh2[tid] = s2; __syncthreads();
    for (int o = 128; o > 0; o >>= 1) {
        if (tid < o) { sh[tid] += sh[tid + o]; sh2[tid] += sh2[tid + o]; }
        __syncthreads();
    }
    float mean = sh[0] * (1.f / H);
    float var  = sh2[0] * (1.f / H) - mean * mean;
    float inv  = rsqrtf(var + EPSLN);
    float acc = 0.f;
    for (int c = tid; c < H; c += 256)
        acc += ((xr[c] - mean) * inv * g[c] + bb[c]) * w[c];
    __syncthreads();
    sh[tid] = acc; __syncthreads();
    for (int o = 128; o > 0; o >>= 1) {
        if (tid < o) sh[tid] += sh[tid + o];
        __syncthreads();
    }
    if (tid == 0) out[row] = sh[0] + ob[0];
}

// ---------------------------------------------------------------- launcher
extern "C" void kernel_launch(void* const* d_in, const int* in_sizes, int n_in,
                              void* d_out, int out_size, void* d_ws, size_t ws_size,
                              hipStream_t stream) {
    const float* nf     = (const float*)d_in[0];
    const int*   batch  = (const int*)  d_in[1];
    const float* cq     = (const float*)d_in[2];
    const float* sa_wq = (const float*)d_in[3],  *sa_bq = (const float*)d_in[4];
    const float* sa_wk = (const float*)d_in[5],  *sa_bk = (const float*)d_in[6];
    const float* sa_wv = (const float*)d_in[7],  *sa_bv = (const float*)d_in[8];
    const float* sa_wo = (const float*)d_in[9],  *sa_bo = (const float*)d_in[10];
    const float* ca_wq = (const float*)d_in[11], *ca_bq = (const float*)d_in[12];
    const float* ca_wk = (const float*)d_in[13], *ca_bk = (const float*)d_in[14];
    const float* ca_wv = (const float*)d_in[15], *ca_bv = (const float*)d_in[16];
    const float* ca_wo = (const float*)d_in[17], *ca_bo = (const float*)d_in[18];
    const float* ln1g = (const float*)d_in[19], *ln1b = (const float*)d_in[20];
    const float* ln2g = (const float*)d_in[21], *ln2b = (const float*)d_in[22];
    const float* ln3g = (const float*)d_in[23], *ln3b = (const float*)d_in[24];
    const float* ffw1 = (const float*)d_in[25], *ffb1 = (const float*)d_in[26];
    const float* ffw2 = (const float*)d_in[27], *ffb2 = (const float*)d_in[28];
    const float* olng = (const float*)d_in[29], *olnb = (const float*)d_in[30];
    const float* outw = (const float*)d_in[31], *outb = (const float*)d_in[32];

    char* wp = (char*)d_ws;
    auto alloc = [&](size_t bytes) -> void* {
        void* r = (void*)wp;
        wp += (bytes + 255) & ~(size_t)255;
        return r;
    };

    int* offs  = (int*)alloc((BG + 1) * 4);
    int* cnt   = (int*)alloc(BG * 4);
    int* aoff  = (int*)alloc((BG + 1) * 4);
    int* sbase = (int*)alloc(BG * 4);
    int* scnt  = (int*)alloc(BG * 4);
    int* rmap  = (int*)alloc((size_t)NNODE * 4);

    bf16* Xn  = (bf16*)alloc((size_t)NNODE * H * 2);
    bf16* WqS = (bf16*)alloc((size_t)H * H * 2);
    bf16* WkS = (bf16*)alloc((size_t)H * H * 2);
    bf16* WvS = (bf16*)alloc((size_t)H * H * 2);
    bf16* WoS = (bf16*)alloc((size_t)H * H * 2);
    bf16* WqC = (bf16*)alloc((size_t)H * H * 2);
    bf16* WkC = (bf16*)alloc((size_t)H * H * 2);
    bf16* WvC = (bf16*)alloc((size_t)H * H * 2);
    bf16* WoC = (bf16*)alloc((size_t)H * H * 2);
    bf16* W1t = (bf16*)alloc((size_t)FF * H * 2);
    bf16* W2t = (bf16*)alloc((size_t)H * FF * 2);

    float* x    = (float*)alloc((size_t)RQ * H * 4);
    bf16*  y    = (bf16*) alloc((size_t)RQ * H * 2);
    bf16*  Qb   = (bf16*) alloc((size_t)RQ * H * 2);
    bf16*  Kb   = (bf16*) alloc((size_t)RQ * H * 2);
    bf16*  Vtb  = (bf16*) alloc((size_t)H * RQ * 2);
    bf16*  Kc   = (bf16*) alloc((size_t)NPAD * H * 2);
    bf16*  Vct  = (bf16*) alloc((size_t)H * NPAD * 2);
    float* aO   = (float*)alloc((size_t)RQ * H * 4);
    bf16*  aObf = (bf16*) alloc((size_t)RQ * H * 2);
    bf16*  ffh  = (bf16*) alloc((size_t)RQ * FF * 2);
    (void)ws_size; (void)n_in; (void)in_sizes; (void)out_size;

    auto cvt = [&](const float* in, bf16* out, long n) {
        k_cvt<<<(unsigned)((n + 255) / 256), 256, 0, stream>>>(in, out, n);
    };
    auto cvtT = [&](const float* W, bf16* Wt, int K, int Nc) {
        long n = (long)K * Nc;
        k_cvtT<<<(unsigned)((n + 255) / 256), 256, 0, stream>>>(W, Wt, K, Nc);
    };

    // --- setup
    k_offsets<<<1, 128, 0, stream>>>(batch, NNODE, offs, cnt, aoff, sbase, scnt);
    k_rmap<<<(NNODE + 255) / 256, 256, 0, stream>>>(batch, NNODE, offs, aoff, rmap);
    cvt(nf, Xn, (long)NNODE * H);
    cvtT(sa_wq, WqS, H, H); cvtT(sa_wk, WkS, H, H); cvtT(sa_wv, WvS, H, H); cvtT(sa_wo, WoS, H, H);
    cvtT(ca_wq, WqC, H, H); cvtT(ca_wk, WkC, H, H); cvtT(ca_wv, WvC, H, H); cvtT(ca_wo, WoC, H, H);
    cvtT(ffw1, W1t, H, FF); cvtT(ffw2, W2t, FF, H);
    k_bcast<<<(unsigned)(((long)RQ * H + 255) / 256), 256, 0, stream>>>(cq, x);

    dim3 gQ(H / 64, RQ / 128);      // 4096xH GEMMs   (128-row block tiles)
    dim3 gN(H / 64, NNODE / 128);   // node K/V projections (32000/128 = 250)
    dim3 gF1(FF / 64, RQ / 128);    // FFN up
    dim3 gA(2, NH, BG);             // attention

    // --- self-attention block
    k_ln<<<RQ, 256, 0, stream>>>(x, ln1g, ln1b, y);
    k_gemm<1, false><<<gQ, 128, 0, stream>>>(y, WqS, sa_bq, nullptr, Qb,  nullptr, RQ, H, H, 0);
    k_gemm<1, false><<<gQ, 128, 0, stream>>>(y, WkS, sa_bk, nullptr, Kb,  nullptr, RQ, H, H, 0);
    k_gemm<2, false><<<gQ, 128, 0, stream>>>(y, WvS, sa_bv, nullptr, Vtb, nullptr, RQ, H, H, RQ);
    k_attn<<<gA, 32, 0, stream>>>(Qb, Kb, Vtb, sbase, scnt, aO, RQ);
    cvt(aO, aObf, (long)RQ * H);
    k_gemm<0, false><<<gQ, 128, 0, stream>>>(aObf, WoS, sa_bo, x, x, nullptr, RQ, H, H, 0);

    // --- cross-attention block
    k_ln<<<RQ, 256, 0, stream>>>(x, ln2g, ln2b, y);
    k_gemm<1, false><<<gQ, 128, 0, stream>>>(y, WqC, ca_bq, nullptr, Qb, nullptr, RQ, H, H, 0);
    k_gemm<1, false><<<gN, 128, 0, stream>>>(Xn, WkC, ca_bk, nullptr, Kc,  rmap, NNODE, H, H, 0);
    k_gemm<2, false><<<gN, 128, 0, stream>>>(Xn, WvC, ca_bv, nullptr, Vct, rmap, NNODE, H, H, NPAD);
    k_attn<<<gA, 32, 0, stream>>>(Qb, Kc, Vct, aoff, cnt, aO, NPAD);
    cvt(aO, aObf, (long)RQ * H);
    k_gemm<0, false><<<gQ, 128, 0, stream>>>(aObf, WoC, ca_bo, x, x, nullptr, RQ, H, H, 0);

    // --- FFN block
    k_ln<<<RQ, 256, 0, stream>>>(x, ln3g, ln3b, y);
    k_gemm<1, true ><<<gF1, 128, 0, stream>>>(y, W1t, ffb1, nullptr, ffh, nullptr, RQ, FF, H, 0);
    k_gemm<0, false><<<gQ,  128, 0, stream>>>(ffh, W2t, ffb2, x, x, nullptr, RQ, H, FF, 0);

    // --- output head
    k_out<<<RQ, 256, 0, stream>>>(x, olng, olnb, outw, outb, (float*)d_out);
}
